// GCNConvWithNormAndDropout_66245575574019
// MI455X (gfx1250) — matrix operand order repaired and checked by reading, not compiled, not run on hardware
//
#include <hip/hip_runtime.h>
#include <hip/hip_bf16.h>
#include <stdint.h>

typedef __attribute__((ext_vector_type(16))) __bf16 v16bf;
typedef __attribute__((ext_vector_type(2)))  __bf16 v2bf;
typedef __attribute__((ext_vector_type(8)))  float  v8f;

#define F_IN   128
#define F_HID  128
#define F_OUT  40
#define BN_EPS 1e-5f

// Hardware bf16 conversion (v_cvt_pk_bf16_f32 on gfx1250)
__device__ __forceinline__ unsigned short f2bf(float f) {
    return __builtin_bit_cast(unsigned short, (__bf16)f);
}
__device__ __forceinline__ unsigned int f2bf2(float a, float b) {
    v2bf t; t[0] = (__bf16)a; t[1] = (__bf16)b;
    return __builtin_bit_cast(unsigned int, t);
}

// ---------------------------------------------------------------- utilities
__global__ void k_zero(float* __restrict__ p, int n) {
    int i = blockIdx.x * blockDim.x + threadIdx.x;
    if (i < n) p[i] = 0.0f;
}

__global__ void k_deg(const int* __restrict__ dst, float* __restrict__ deg, int E) {
    int e = blockIdx.x * blockDim.x + threadIdx.x;
    if (e < E) atomicAdd(&deg[dst[e]], 1.0f);
}

__global__ void k_dinv(float* __restrict__ deg, int n) {
    int i = blockIdx.x * blockDim.x + threadIdx.x;
    if (i < n) {
        float d = deg[i] + 1.0f;                       // + self loop
        deg[i] = (d > 0.0f) ? rsqrtf(d) : 0.0f;        // in place: deg -> dinv
    }
}

// ------------------------------------------------- WMMA GEMM: C = A(f32)@W(f32)
// A: [nrows x 128] row-major, W: [128 x COLS_REAL] row-major, C: [nrows x COLS_REAL]
// Block = 256 threads (8 waves), covers 128 rows; each wave owns a 16-row strip.
template<int COLS_PAD, int COLS_REAL>
__global__ void __launch_bounds__(256)
k_gemm(const float* __restrict__ A, const float* __restrict__ W,
       float* __restrict__ C, int nrows) {
    constexpr int KP = 136;                            // padded K stride (bf16 elems)
    __shared__ __align__(16) unsigned short Wl[COLS_PAD * KP];

    // cooperative transpose-load of W into LDS as bf16: Wl[col][k]
    for (int idx = threadIdx.x; idx < 128 * COLS_PAD; idx += 256) {
        int k = idx / COLS_PAD;
        int n = idx - k * COLS_PAD;
        float v = (n < COLS_REAL) ? W[k * COLS_REAL + n] : 0.0f;
        Wl[n * KP + k] = f2bf(v);
    }
    __syncthreads();

    const int lane = threadIdx.x & 31;
    const int wave = threadIdx.x >> 5;
    const int l16  = lane & 15;
    const int hsel = lane >> 4;                        // 0 or 1
    const int h8   = hsel * 8;
    const int rowBase = blockIdx.x * 128 + wave * 16;
    const bool fullTile = (rowBase + 16 <= nrows);     // wave-uniform fast path

    int arow = rowBase + l16;                          // clamp (keep EXEC all-1 for WMMA)
    if (arow > nrows - 1) arow = nrows - 1;
    const float* ap = A + (size_t)arow * 128;

    // A fragments: 16x32 bf16 per 32-K step.
    // lanes 0-15: K = {kk*32+0..7, kk*32+16..23}; lanes 16-31: +8 on both runs.
    union AV { v16bf v; unsigned int u32[8]; } afr[4];
#pragma unroll
    for (int kk = 0; kk < 4; ++kk) {
        const float* p = ap + kk * 32;
        float4 f0 = *(const float4*)(p + h8);
        float4 f1 = *(const float4*)(p + h8 + 4);
        float4 f2 = *(const float4*)(p + 16 + h8);
        float4 f3 = *(const float4*)(p + 20 + h8);
        afr[kk].u32[0] = f2bf2(f0.x, f0.y);
        afr[kk].u32[1] = f2bf2(f0.z, f0.w);
        afr[kk].u32[2] = f2bf2(f1.x, f1.y);
        afr[kk].u32[3] = f2bf2(f1.z, f1.w);
        afr[kk].u32[4] = f2bf2(f2.x, f2.y);
        afr[kk].u32[5] = f2bf2(f2.z, f2.w);
        afr[kk].u32[6] = f2bf2(f3.x, f3.y);
        afr[kk].u32[7] = f2bf2(f3.z, f3.w);
    }

    constexpr int NT = COLS_PAD / 16;
#pragma unroll
    for (int ct = 0; ct < NT; ++ct) {
        v8f acc = {0.f, 0.f, 0.f, 0.f, 0.f, 0.f, 0.f, 0.f};
        const int col = ct * 16 + l16;
        // B fragment: lanes 0-15 K=kk*32+{0..7,16..23}, lanes 16-31 +8 on both runs, col = l16
        const unsigned short* bbase = &Wl[col * KP + hsel * 8];
#pragma unroll
        for (int kk = 0; kk < 4; ++kk) {
            union BV { v16bf v; uint4 q[2]; } bf;
            const unsigned short* bp = bbase + kk * 32;
            bf.q[0] = *(const uint4*)(bp);
            bf.q[1] = *(const uint4*)(bp + 16);
            acc = __builtin_amdgcn_wmma_f32_16x16x32_bf16(
                false, afr[kk].v, false, bf.v, (short)0, acc, false, false);
        }
        // store epilogue: C layout -> lane holds column `col`, rows rowBase+h8+r
        if (col < COLS_REAL) {
            float* cp = C + (size_t)(rowBase + h8) * COLS_REAL + col;
            if (fullTile) {
#pragma unroll
                for (int r = 0; r < 8; ++r)            // immediate-offset stores
                    cp[(size_t)r * COLS_REAL] = acc[r];
            } else {
#pragma unroll
                for (int r = 0; r < 8; ++r) {
                    int m = rowBase + h8 + r;
                    if (m < nrows) C[(size_t)m * COLS_REAL + col] = acc[r];
                }
            }
        }
    }
}

// ------------------------------------------ self-loop init: out = dinv^2 * h
template<int F>
__global__ void k_selfloop(const float* __restrict__ h, const float* __restrict__ dinv,
                           float* __restrict__ out, int n) {
    int i = blockIdx.x * blockDim.x + threadIdx.x;     // over n*F/4 float4 chunks
    if (i >= n * (F / 4)) return;
    int node = i / (F / 4);
    float dv = dinv[node];
    float s = dv * dv;
    const float4 v = *(const float4*)(h + (size_t)i * 4);
    float4 o; o.x = s * v.x; o.y = s * v.y; o.z = s * v.z; o.w = s * v.w;
    *(float4*)(out + (size_t)i * 4) = o;
}

// --------------------------------- edge scatter: out[d] += dinv[s]*dinv[d]*h[s]
template<int CH, int F>
__global__ void k_edge(const float* __restrict__ h, const int* __restrict__ src,
                       const int* __restrict__ dst, const float* __restrict__ dinv,
                       float* __restrict__ out, int E) {
    int i = blockIdx.x * blockDim.x + threadIdx.x;     // over E*CH
    if (i >= E * CH) return;
    int e = i / CH;
    int c = i - e * CH;
    int s = src[e], d = dst[e];
    float nrm = dinv[s] * dinv[d];
    const float4 v = *(const float4*)(h + (size_t)s * F + c * 4);
    float* o = out + (size_t)d * F + c * 4;
    atomicAdd(o + 0, nrm * v.x);
    atomicAdd(o + 1, nrm * v.y);
    atomicAdd(o + 2, nrm * v.z);
    atomicAdd(o + 3, nrm * v.w);
}

// ---------------------------------------------------------- BatchNorm stats
__global__ void k_bnstats(const float* __restrict__ h, const float* __restrict__ b1,
                          float* __restrict__ sums, float* __restrict__ sumsq, int n) {
    int f = threadIdx.x;                               // 128 threads
    float bb = b1[f];
    float s = 0.f, q = 0.f;
    for (int r = blockIdx.x; r < n; r += gridDim.x) {
        float v = h[(size_t)r * 128 + f] + bb;
        s += v; q += v * v;
    }
    atomicAdd(&sums[f], s);
    atomicAdd(&sumsq[f], q);
}

__global__ void k_bnnorm(float* __restrict__ h, const float* __restrict__ b1,
                         const float* __restrict__ gamma, const float* __restrict__ beta,
                         const float* __restrict__ sums, const float* __restrict__ sumsq,
                         int n) {
    int i = blockIdx.x * blockDim.x + threadIdx.x;     // over n*128
    if (i >= n * 128) return;
    int f = i & 127;
    float inv_n = 1.0f / (float)n;
    float mean = sums[f] * inv_n;
    float var  = sumsq[f] * inv_n - mean * mean;
    float v = h[(size_t)i] + b1[f];
    v = (v - mean) * rsqrtf(var + BN_EPS) * gamma[f] + beta[f];
    h[(size_t)i] = fmaxf(v, 0.0f);                     // ReLU (dropout = identity)
}

// --------------------------------------------------- log_softmax over 40 cols
__global__ void k_lsm(float* __restrict__ out, const float* __restrict__ b2, int n) {
    int lane = threadIdx.x & 31;
    int node = blockIdx.x * (blockDim.x >> 5) + (threadIdx.x >> 5);
    if (node >= n) return;
    float* row = out + (size_t)node * F_OUT;
    float v0 = row[lane] + b2[lane];                   // lane < 32 <= 40 always valid
    int f1 = lane + 32;
    float v1 = (f1 < F_OUT) ? row[f1] + b2[f1] : -__builtin_inff();
    float m = fmaxf(v0, v1);
    for (int off = 16; off; off >>= 1) m = fmaxf(m, __shfl_xor(m, off, 32));
    float s = expf(v0 - m) + ((f1 < F_OUT) ? expf(v1 - m) : 0.0f);
    for (int off = 16; off; off >>= 1) s += __shfl_xor(s, off, 32);
    float lse = m + logf(s);
    row[lane] = v0 - lse;
    if (f1 < F_OUT) row[f1] = v1 - lse;
}

// =========================================================================
extern "C" void kernel_launch(void* const* d_in, const int* in_sizes, int n_in,
                              void* d_out, int out_size, void* d_ws, size_t ws_size,
                              hipStream_t stream) {
    const float* x     = (const float*)d_in[0];
    const float* W1    = (const float*)d_in[1];
    const float* b1    = (const float*)d_in[2];
    const float* gamma = (const float*)d_in[3];
    const float* beta  = (const float*)d_in[4];
    const float* W2    = (const float*)d_in[5];
    const float* b2    = (const float*)d_in[6];
    const int*   esrc  = (const int*)d_in[7];
    const int*   edst  = (const int*)d_in[8];

    const int n = in_sizes[0] / F_IN;                  // 100000
    const int E = in_sizes[7];                         // 1600000
    float* out = (float*)d_out;

    // workspace layout (floats): dinv[n] | sums[128] | sumsq[128] | H1[n*128] | Hagg[n*128] | H2[n*40]
    float* dinv  = (float*)d_ws;
    float* sums  = dinv + n;
    float* sumsq = sums + 128;
    float* H1    = sumsq + 128;
    float* Hagg  = H1 + (size_t)n * F_HID;
    float* H2    = Hagg + (size_t)n * F_HID;

    auto cdiv = [](long a, long b) { return (int)((a + b - 1) / b); };

    // degrees + norm
    k_zero<<<cdiv(n + 256, 256), 256, 0, stream>>>(dinv, n + 256);     // dinv+sums+sumsq
    k_deg<<<cdiv(E, 256), 256, 0, stream>>>(edst, dinv, E);
    k_dinv<<<cdiv(n, 256), 256, 0, stream>>>(dinv, n);

    // layer 1: H1 = x @ W1 ; Hagg = D^-1/2 A D^-1/2 H1 ; BN ; ReLU
    k_gemm<128, 128><<<cdiv(n, 128), 256, 0, stream>>>(x, W1, H1, n);
    k_selfloop<128><<<cdiv((long)n * 32, 256), 256, 0, stream>>>(H1, dinv, Hagg, n);
    k_edge<32, 128><<<cdiv((long)E * 32, 256), 256, 0, stream>>>(H1, esrc, edst, dinv, Hagg, E);
    k_bnstats<<<512, 128, 0, stream>>>(Hagg, b1, sums, sumsq, n);
    k_bnnorm<<<cdiv((long)n * 128, 256), 256, 0, stream>>>(Hagg, b1, gamma, beta, sums, sumsq, n);

    // layer 2: H2 = Hagg @ W2 ; out = D^-1/2 A D^-1/2 H2 + b2 ; log_softmax
    k_gemm<48, F_OUT><<<cdiv(n, 128), 256, 0, stream>>>(Hagg, W2, H2, n);
    k_selfloop<F_OUT><<<cdiv((long)n * 10, 256), 256, 0, stream>>>(H2, dinv, out, n);
    k_edge<10, F_OUT><<<cdiv((long)E * 10, 256), 256, 0, stream>>>(H2, esrc, edst, dinv, out, E);
    k_lsm<<<cdiv(n, 8), 256, 0, stream>>>(out, b2, n);
}
